// BiAttention_50697793962351
// MI455X (gfx1250) — matrix operand order repaired and checked by reading, not compiled, not run on hardware
//
#include <hip/hip_runtime.h>
#include <hip/hip_bf16.h>

#define BB 16
#define NN 1024
#define MM 1024
#define DD 128
#define VERY_NEG (-1e30f)

typedef __attribute__((ext_vector_type(16))) __bf16 v16bf;
typedef __attribute__((ext_vector_type(8)))  __bf16 v8bf;
typedef __attribute__((ext_vector_type(4)))  __bf16 v4bf;
typedef __attribute__((ext_vector_type(8)))  float  v8f;

// ---------------- LDS layout for k1 ----------------
#define MT_STRIDE 136                    // bf16 elems per Mtile row (128 + 8 pad)
#define MT_BYTES  (32 * MT_STRIDE * 2)   // 8704 B per buffer
#define TT_STRIDE 40                     // bf16 elems per MtileT row (32 + 8 pad)
#define TT_BYTES  (128 * TT_STRIDE * 2)  // 10240 B per buffer
#define TT_OFF    (2 * MT_BYTES)         // 17408
#define TERM_OFF  (TT_OFF + 2 * TT_BYTES)// 37888 (16B aligned)
#define LDS_BYTES (TERM_OFF + 2 * 256)   // 38400
#define OS_STRIDE 132                    // f32 per row in out-staging (overlay)
#define OS_ELEMS  (16 * OS_STRIDE)       // 8448 B per wave slot

// ------- CDNA5 async copy helpers (ASYNCcnt-tracked LDS DMA) -------
__device__ __forceinline__ unsigned lds_off(const void* p) {
  // low 32 bits of a generic shared-memory address == LDS byte offset
  return (unsigned)(size_t)p;
}
__device__ __forceinline__ void async_b128(unsigned lds, unsigned goff,
                                           const void* base) {
  asm volatile("global_load_async_to_lds_b128 %0, %1, %2"
               :: "v"(lds), "v"(goff), "s"(base)
               : "memory");
}
__device__ __forceinline__ void wait_async0() {
  asm volatile("s_wait_asynccnt 0" ::: "memory");
}

__device__ __forceinline__ v16bf cat8(v8bf lo, v8bf hi) {
  v16bf r;
#pragma unroll
  for (int e = 0; e < 8; ++e) { r[e] = lo[e]; r[e + 8] = hi[e]; }
  return r;
}

// ---- k0: term[b*M+m] = {dot(memory[b,m,:], w_m), mm ? 0 : -1e30} ----
__global__ __launch_bounds__(256) void k0_term(const float* __restrict__ memory,
                                               const float* __restrict__ w_m,
                                               const int* __restrict__ memory_mask,
                                               float2* __restrict__ term) {
  const int row = blockIdx.x * 16 + (threadIdx.x >> 4);
  const int j = threadIdx.x & 15;
  const float* mp = memory + (size_t)row * DD + j * 8;
  float s = 0.f;
#pragma unroll
  for (int e = 0; e < 8; ++e) s += mp[e] * w_m[j * 8 + e];
  s += __shfl_xor(s, 8, 32);
  s += __shfl_xor(s, 4, 32);
  s += __shfl_xor(s, 2, 32);
  s += __shfl_xor(s, 1, 32);
  if (j == 0) term[row] = make_float2(s, memory_mask[row] ? 0.f : VERY_NEG);
}

// ---- k0b: memory f32 -> bf16 in row-major (memB) and d-major (memT) ----
__global__ __launch_bounds__(256) void k0b_convert(const float* __restrict__ memory,
                                                   __bf16* __restrict__ memB,
                                                   __bf16* __restrict__ memT) {
  __shared__ __align__(16) __bf16 tr[128][136];  // [d][m-local], padded
  const int tid = threadIdx.x;
  const int b = blockIdx.x >> 3;
  const int m0 = (blockIdx.x & 7) * 128;
#pragma unroll
  for (int i = 0; i < 16; ++i) {
    const int fi = i * 256 + tid;  // 4096 float4 = 128 rows x 128 d
    const int row = fi >> 5;
    const int dc = (fi & 31) * 4;
    const float4 v =
        *(const float4*)(memory + (size_t)(b * MM + m0 + row) * DD + dc);
    const __bf16 c0 = (__bf16)v.x, c1 = (__bf16)v.y, c2 = (__bf16)v.z,
                 c3 = (__bf16)v.w;
    v4bf pk; pk[0] = c0; pk[1] = c1; pk[2] = c2; pk[3] = c3;
    *(v4bf*)(memB + (size_t)(b * MM + m0 + row) * DD + dc) = pk;
    tr[dc + 0][row] = c0;
    tr[dc + 1][row] = c1;
    tr[dc + 2][row] = c2;
    tr[dc + 3][row] = c3;
  }
  __syncthreads();
#pragma unroll
  for (int i = 0; i < 8; ++i) {
    const int wi = i * 256 + tid;  // 2048 16B chunks = 128 d x 16 chunks
    const int d = wi >> 4;
    const int ch = wi & 15;
    const v8bf v = *(const v8bf*)(&tr[d][ch * 8]);
    *(v8bf*)(memT + (size_t)(b * DD + d) * MM + m0 + ch * 8) = v;
  }
}

// ---------------- k1: fused trilinear attention (flash style) ----------------
__global__ __launch_bounds__(256) void k1_attn(
    const float* __restrict__ query, const float* __restrict__ w_q,
    const float* __restrict__ w_qm, const int* __restrict__ query_mask,
    const __bf16* __restrict__ memB, const __bf16* __restrict__ memT,
    const float2* __restrict__ term, float* __restrict__ out1,
    float* __restrict__ rowmax_ws) {
  __shared__ __align__(16) char smem[LDS_BYTES];
  const int tid = threadIdx.x;
  const int w = tid >> 5;
  const int lane = tid & 31;
  const int l16 = lane & 15;
  const int h = lane >> 4;
  const int b = blockIdx.x >> 3;
  const int q0 = (blockIdx.x & 7) * 128;
  const int qrow = q0 + w * 16 + l16;  // this lane's q column

  // ---- load Qw = q * w_qm into B-operand fragments (registers) + qb ----
  float qb = 0.f;
  v16bf Qw[4];
#pragma unroll
  for (int c = 0; c < 4; ++c) {
    const int based = 32 * c + 16 * h;  // B layout: lane half h holds K = 16h..16h+15
    const float* qp = query + (size_t)(b * NN + qrow) * DD + based;
    const float* aq = w_q + based;
    const float* am = w_qm + based;
#pragma unroll
    for (int e = 0; e < 16; ++e) {
      const float qv = qp[e];
      qb += qv * aq[e];
      Qw[c][e] = (__bf16)(qv * am[e]);
    }
  }
  qb += __shfl_xor(qb, 16, 32);
  const float qmf = (float)query_mask[b * NN + qrow];

  // ---- async (ASYNCcnt) staging of the next m-tile: pure LDS DMA ----
  auto stage_async = [&](int step, int bb) {
    const int m0 = step * 32;
    char* mtB = smem + bb * MT_BYTES;
    char* ttB = smem + TT_OFF + bb * TT_BYTES;
#pragma unroll
    for (int j = 0; j < 2; ++j) {  // 2 instrs x 2 rows x 256B : rows 4w..4w+3
      const int row = w * 4 + j * 2 + h;
      async_b128(lds_off(mtB + row * (MT_STRIDE * 2) + l16 * 16),
                 (unsigned)(((b * MM + m0 + row) * DD + l16 * 8) * 2), memB);
    }
#pragma unroll
    for (int j = 0; j < 2; ++j) {  // 2 instrs x 8 d-rows x 64B
      const int dr = (w * 2 + j) * 8 + (lane >> 2);
      const int ch = lane & 3;
      async_b128(lds_off(ttB + dr * (TT_STRIDE * 2) + ch * 16),
                 (unsigned)(((b * DD + dr) * MM + m0 + ch * 8) * 2), memT);
    }
    if (w == 0) {  // 32 float2 term values (256B)
      const int c = lane & 15;
      async_b128(lds_off(smem + TERM_OFF + bb * 256 + c * 16),
                 (unsigned)((b * MM + m0) * 8 + c * 16), term);
    }
  };

  v8f Ot[8];  // O^T accumulator: tile t, vgpr r -> d = 16t+r+8h, q = l16
#pragma unroll
  for (int t = 0; t < 8; ++t)
#pragma unroll
    for (int r = 0; r < 8; ++r) Ot[t][r] = 0.f;
  float run_max = -3.0e38f, run_sum = 0.f;

  stage_async(0, 0);
  wait_async0();
  __syncthreads();

  for (int step = 0; step < 32; ++step) {
    const int bb = step & 1;
    if (step + 1 < 32) stage_async(step + 1, bb ^ 1);

    __bf16* mt = (__bf16*)(smem + bb * MT_BYTES);
    __bf16* tt = (__bf16*)(smem + TT_OFF + bb * TT_BYTES);
    const float2* tp = (const float2*)(smem + TERM_OFF + bb * 256);

    // ---- S^T = Mem_tile @ Qw^T : two 16x16 tiles (m 0-15, m 16-31) ----
    v8f sA = {0, 0, 0, 0, 0, 0, 0, 0}, sB = {0, 0, 0, 0, 0, 0, 0, 0};
#pragma unroll
    for (int c = 0; c < 4; ++c) {
      const int off = 32 * c + 16 * h;  // A layout: K runs [16h,16h+16)
      const v16bf aA = cat8(*(v8bf*)(mt + l16 * MT_STRIDE + off),
                            *(v8bf*)(mt + l16 * MT_STRIDE + off + 8));
      sA = __builtin_amdgcn_wmma_f32_16x16x32_bf16(false, aA, false, Qw[c],
                                                   (short)0, sA, false, false);
      const v16bf aB = cat8(*(v8bf*)(mt + (16 + l16) * MT_STRIDE + off),
                            *(v8bf*)(mt + (16 + l16) * MT_STRIDE + off + 8));
      sB = __builtin_amdgcn_wmma_f32_16x16x32_bf16(false, aB, false, Qw[c],
                                                   (short)0, sB, false, false);
    }

    // ---- online softmax over the 32 m values (per q = l16) ----
    float pA[8], pB[8];
    float tmax = -3.0e38f;
#pragma unroll
    for (int r = 0; r < 8; ++r) {
      const float2 tA = tp[r + 8 * h];
      const float2 tB = tp[16 + r + 8 * h];
      const float t0 = sA[r] + fmaf(qmf, tA.y, tA.x);
      const float t1 = sB[r] + fmaf(qmf, tB.y, tB.x);
      pA[r] = t0;
      pB[r] = t1;
      tmax = fmaxf(tmax, fmaxf(t0, t1));
    }
    tmax = fmaxf(tmax, __shfl_xor(tmax, 16, 32));
    const float nm = fmaxf(run_max, tmax);
    const float sc = __expf(run_max - nm);
    run_max = nm;
    float ts = 0.f;
#pragma unroll
    for (int r = 0; r < 8; ++r) {
      pA[r] = __expf(pA[r] - nm); ts += pA[r];
      pB[r] = __expf(pB[r] - nm); ts += pB[r];
    }
    ts += __shfl_xor(ts, 16, 32);
    run_sum = run_sum * sc + ts;
#pragma unroll
    for (int t = 0; t < 8; ++t)
#pragma unroll
      for (int r = 0; r < 8; ++r) Ot[t][r] *= sc;

    // ---- convert P^T (C layout) to B-operand layout via half swap ----
    float pAs[8], pBs[8];
#pragma unroll
    for (int r = 0; r < 8; ++r) {
      pAs[r] = __shfl_xor(pA[r], 16, 32);
      pBs[r] = __shfl_xor(pB[r], 16, 32);
    }
    v16bf Pv;
#pragma unroll
    for (int e = 0; e < 8; ++e) {
      Pv[e]     = (__bf16)(h == 0 ? pA[e] : pBs[e]);   // K = 16h + e
      Pv[e + 8] = (__bf16)(h == 0 ? pAs[e] : pB[e]);   // K = 16h + 8 + e
    }

    // ---- O^T += Mem^T @ P^T over 8 d-tiles ----
#pragma unroll
    for (int t = 0; t < 8; ++t) {
      __bf16* tr = tt + (16 * t + l16) * TT_STRIDE;
      const v16bf aT = cat8(*(v8bf*)(tr + 16 * h), *(v8bf*)(tr + 16 * h + 8));
      Ot[t] = __builtin_amdgcn_wmma_f32_16x16x32_bf16(false, aT, false, Pv,
                                                      (short)0, Ot[t], false,
                                                      false);
    }
    wait_async0();   // next tile's DMA complete before the barrier
    __syncthreads();
  }

  const float inv = 1.f / run_sum;
  if (lane < 16)
    rowmax_ws[b * NN + q0 + w * 16 + lane] =
        run_max + qb + (qmf == 0.f ? VERY_NEG : 0.f);

  // ---- transpose O^T through LDS for coalesced stores (2 rounds, 4 waves) ----
  float* osBase = (float*)smem;
  for (int round = 0; round < 2; ++round) {
    if ((w >> 2) == round) {
      float* os = osBase + (w & 3) * OS_ELEMS;
#pragma unroll
      for (int t = 0; t < 8; ++t)
#pragma unroll
        for (int r = 0; r < 8; ++r)
          os[l16 * OS_STRIDE + 16 * t + r + 8 * h] = Ot[t][r] * inv;
    }
    __syncthreads();
    if ((w >> 2) == round) {
      float* os = osBase + (w & 3) * OS_ELEMS;
#pragma unroll
      for (int q = 0; q < 16; ++q) {
        const float4 v = *(float4*)(os + q * OS_STRIDE + 4 * lane);
        *(float4*)(out1 + (size_t)(b * NN + q0 + w * 16 + q) * DD + 4 * lane) =
            v;
      }
    }
    __syncthreads();
  }
}

// ---------------- k2: partial m2q = softmax(rowmax) @ query ----------------
__global__ __launch_bounds__(256) void k2_m2q(const float* __restrict__ query,
                                              const float* __restrict__ rowmax_ws,
                                              float* __restrict__ partial) {
  __shared__ float srm[1024];
  __shared__ float red[256];
  __shared__ float sacc[128];
  const int tid = threadIdx.x;
  const int b = blockIdx.x >> 2;
  const int g = blockIdx.x & 3;
#pragma unroll
  for (int i = 0; i < 4; ++i)
    srm[i * 256 + tid] = rowmax_ws[b * NN + i * 256 + tid];
  __syncthreads();
  float tmax = -3.0e38f;
#pragma unroll
  for (int i = 0; i < 4; ++i) tmax = fmaxf(tmax, srm[i * 256 + tid]);
  red[tid] = tmax;
  __syncthreads();
  for (int s = 128; s > 0; s >>= 1) {
    if (tid < s) red[tid] = fmaxf(red[tid], red[tid + s]);
    __syncthreads();
  }
  const float bmax = red[0];
  __syncthreads();
  float tsum = 0.f;
#pragma unroll
  for (int i = 0; i < 4; ++i) tsum += __expf(srm[i * 256 + tid] - bmax);
  red[tid] = tsum;
  __syncthreads();
  for (int s = 128; s > 0; s >>= 1) {
    if (tid < s) red[tid] += red[tid + s];
    __syncthreads();
  }
  const float inv = 1.f / red[0];
  __syncthreads();
  const int half = tid >> 7;
  const int d = tid & 127;
  const int n0 = g * 256 + half * 128;
  float acc = 0.f;
  for (int n = n0; n < n0 + 128; ++n)
    acc += __expf(srm[n] - bmax) * inv * query[(size_t)(b * NN + n) * DD + d];
  if (half == 0) sacc[d] = acc;
  __syncthreads();
  if (half == 1) sacc[d] += acc;
  __syncthreads();
  if (tid < 128) partial[(b * 4 + g) * 128 + tid] = sacc[tid];
}

// ---------------- k3: broadcast m2q to out2[b, n, :] ----------------
__global__ __launch_bounds__(256) void k3_bcast(const float* __restrict__ partial,
                                                float* __restrict__ out2) {
  const int tid = threadIdx.x;
  const int b = blockIdx.x >> 7;
  const int nc = blockIdx.x & 127;
  const int d0 = (tid & 31) * 4;
  float4 v = {0.f, 0.f, 0.f, 0.f};
#pragma unroll
  for (int gg = 0; gg < 4; ++gg) {
    const float4 p = *(const float4*)(partial + (b * 4 + gg) * 128 + d0);
    v.x += p.x; v.y += p.y; v.z += p.z; v.w += p.w;
  }
  const int n = nc * 8 + (tid >> 5);
  *(float4*)(out2 + (size_t)(b * NN + n) * DD + d0) = v;
}

extern "C" void kernel_launch(void* const* d_in, const int* in_sizes, int n_in,
                              void* d_out, int out_size, void* d_ws,
                              size_t ws_size, hipStream_t stream) {
  (void)in_sizes; (void)n_in; (void)out_size; (void)ws_size;
  const float* query = (const float*)d_in[0];
  const float* memory = (const float*)d_in[1];
  const float* w_q = (const float*)d_in[2];
  const float* w_m = (const float*)d_in[3];
  const float* w_qm = (const float*)d_in[4];
  const int* query_mask = (const int*)d_in[5];
  const int* memory_mask = (const int*)d_in[6];
  float* out1 = (float*)d_out;
  float* out2 = out1 + (size_t)BB * NN * DD;

  char* ws = (char*)d_ws;
  float2* term = (float2*)ws;                                   // 128 KB
  float* rowmax = (float*)(ws + 131072);                        // 64 KB
  float* partial = (float*)(ws + 131072 + 65536);               // 32 KB
  __bf16* memB = (__bf16*)(ws + 229376);                        // 4 MB
  __bf16* memT = (__bf16*)(ws + 229376 + (size_t)BB * MM * DD * 2);

  k0_term<<<BB * MM / 16, 256, 0, stream>>>(memory, w_m, memory_mask, term);
  k0b_convert<<<BB * 8, 256, 0, stream>>>(memory, memB, memT);
  k1_attn<<<BB * (NN / 128), 256, 0, stream>>>(query, w_q, w_qm, query_mask,
                                               memB, memT, term, out1, rowmax);
  k2_m2q<<<BB * 4, 256, 0, stream>>>(query, rowmax, partial);
  k3_bcast<<<BB * (NN / 8), 256, 0, stream>>>(partial, out2);
}
